// SkilledLoRA_65678639890848
// MI455X (gfx1250) — compile-verified, matrix operand order checked by reading, not run on hardware
//
#include <hip/hip_runtime.h>
#include <hip/hip_bf16.h>

typedef _Float16 v8h  __attribute__((ext_vector_type(8)));
typedef _Float16 v16h __attribute__((ext_vector_type(16)));
typedef float    v2f  __attribute__((ext_vector_type(2)));
typedef float    v4f  __attribute__((ext_vector_type(4)));
typedef float    v8f  __attribute__((ext_vector_type(8)));

#define IN_F   2048
#define OUT_F  2048
#define RANK   16
#define BS     8
#define SEQ    2048
#define NROWS  (BS * SEQ)          // 16384
#define MIXN   (BS * IN_F * RANK)  // 262144 elements per mix tensor

// ---------------------------------------------------------------------------
// CDNA5 async global->LDS copy (ASYNCcnt-tracked). The builtin's probed
// signature takes GCC-vector v4i pointers: (v4i as1* src, v4i as3* dst,
// imm offset, imm cpol).
// ---------------------------------------------------------------------------
#if __has_builtin(__builtin_amdgcn_global_load_async_to_lds_b128)
#define HAVE_ASYNC_LDS 1
typedef int v4i_ __attribute__((vector_size(16)));               // matches 'V4i'
typedef v4i_ __attribute__((address_space(1)))* g_v4i_p;         // global ptr
typedef v4i_ __attribute__((address_space(3)))* l_v4i_p;         // LDS ptr
// NOTE: per ISA 10.7/15.18, IOFFSET is added to BOTH the global and LDS
// addresses, so tile layouts below keep identical relative offsets.
#define ALD16(ldsp, gp, off)                                                  \
    __builtin_amdgcn_global_load_async_to_lds_b128(                           \
        (g_v4i_p)(unsigned long long)(gp),                                    \
        (l_v4i_p)(unsigned int)(unsigned long long)(ldsp),                    \
        (off), 0)
#if __has_builtin(__builtin_amdgcn_s_wait_asynccnt)
#define WAIT_ASYNC(n) __builtin_amdgcn_s_wait_asynccnt(n)
#else
#define WAIT_ASYNC(n) asm volatile("s_wait_asynccnt %0" ::"i"(n) : "memory")
#endif
#else
#define ALD16(ldsp, gp, off)                                                  \
    (*(v4f*)((char*)(ldsp) + (off)) = *(const v4f*)((const char*)(gp) + (off)))
#define WAIT_ASYNC(n) ((void)0)
#endif

// ---------------------------------------------------------------------------
// Phase 0: skill-weighted mixing of LoRA factors.
//   Amix[b][i][r] = sum_s weights[b][q][s] * lora_a[q][s][d][r]   (q=i/512,d=i%512)
//   Bmix[b][r][o] = 2.0 * sum_s weights[b][q][s] * lora_b[q][s][r][d]
// ---------------------------------------------------------------------------
__global__ __launch_bounds__(256) void skilled_lora_mix(
    const float* __restrict__ w,      // [8][4][8]
    const float* __restrict__ la,     // [4][8][512][16]
    const float* __restrict__ lb,     // [4][8][16][512]
    float* __restrict__ Amix,         // [8][2048][16]
    float* __restrict__ Bmix)         // [8][16][2048]
{
    int idx = blockIdx.x * blockDim.x + threadIdx.x;
    if (idx >= MIXN) return;

    int b  = idx / (IN_F * RANK);
    int i  = (idx / RANK) % IN_F;
    int r  = idx % RANK;
    int q  = i >> 9;
    int d  = i & 511;
    float accA = 0.f;
    #pragma unroll
    for (int s = 0; s < 8; ++s) {
        accA += w[b * 32 + q * 8 + s] * la[(((q * 8 + s) * 512 + d) << 4) + r];
    }
    Amix[idx] = accA;

    int b2 = idx / (RANK * OUT_F);
    int r2 = (idx / OUT_F) % RANK;
    int o  = idx % OUT_F;
    int q2 = o >> 9;
    int d2 = o & 511;
    float accB = 0.f;
    #pragma unroll
    for (int s = 0; s < 8; ++s) {
        accB += w[b2 * 32 + q2 * 8 + s] * lb[(((q2 * 8 + s) * 16 + r2) << 9) + d2];
    }
    Bmix[idx] = 2.0f * accB;   // fold SCALING here
}

// ---------------------------------------------------------------------------
// Phase 1: T[row][r] = sum_i x[row][i] * Amix[b][i][r]   (row = b*SEQ + t)
// One wave per 16-row tile; v_wmma_f32_16x16x32_f16, K loop over 2048.
// ---------------------------------------------------------------------------
__global__ __launch_bounds__(128) void skilled_lora_xa(
    const float* __restrict__ x,      // [16384][2048]
    const float* __restrict__ Amix,   // [8][2048][16]
    float* __restrict__ T)            // [16384][16]
{
    const int wave = threadIdx.x >> 5;
    const int lane = threadIdx.x & 31;
    const int tile = blockIdx.x * 4 + wave;        // 0..1023
    const int row0 = tile * 16;
    const int b    = row0 >> 11;
    const float* __restrict__ Ab = Amix + (size_t)b * IN_F * RANK;

    const int m  = lane & 15;
    const int kb = (lane >> 4) * 8;

    v8f c = {};
    for (int k0 = 0; k0 < IN_F; k0 += 32) {
        const float* xr = x + (size_t)(row0 + m) * IN_F + k0 + kb;
        v16h a;
        #pragma unroll
        for (int j = 0; j < 8; ++j) {
            a[j]     = (_Float16)xr[j];
            a[j + 8] = (_Float16)xr[16 + j];
        }
        v16h bf;
        #pragma unroll
        for (int j = 0; j < 8; ++j) {
            bf[j]     = (_Float16)Ab[(size_t)(k0 + kb + j) * RANK + m];
            bf[j + 8] = (_Float16)Ab[(size_t)(k0 + 16 + kb + j) * RANK + m];
        }
        c = __builtin_amdgcn_wmma_f32_16x16x32_f16(false, a, false, bf,
                                                   (short)0, c, false, false);
    }
    #pragma unroll
    for (int j = 0; j < 8; ++j) {
        int mrow = ((lane < 16) ? 0 : 8) + j;
        T[(size_t)(row0 + mrow) * RANK + m] = c[j];
    }
}

// ---------------------------------------------------------------------------
// Phase 2: fused  out = x @ W^T + bias + T @ Bmix
// 64x64 block tile, 256 threads (8 waves); wave owns 16x32 of C.
// Double-buffered f32 K-tiles staged via GLOBAL_LOAD_ASYNC_TO_LDS_B128,
// pipelined on ASYNCcnt: issue tile k+1, s_wait_asynccnt 4 => tile k resident.
// ---------------------------------------------------------------------------
__global__ __launch_bounds__(256) void skilled_lora_gemm(
    const float* __restrict__ x,      // [16384][2048]
    const float* __restrict__ W,      // [2048][2048]  (row o, col i)
    const float* __restrict__ bias,   // [2048]
    const float* __restrict__ T,      // [16384][16]
    const float* __restrict__ Bmix,   // [8][16][2048] (scaled)
    float* __restrict__ out)          // [16384][2048]
{
    constexpr int LDF = 36;                        // padded f32 row stride (16B mult)
    __shared__ float xs[2][64 * LDF];
    __shared__ float wl[2][64 * LDF];

    const int tid  = threadIdx.x;
    const int lane = tid & 31;
    const int wave = tid >> 5;
    const int row0 = blockIdx.y * 64;
    const int col0 = blockIdx.x * 64;
    const int b    = row0 >> 11;

    const int mr = wave & 3;                       // row tile within block
    const int nc = wave >> 2;                      // col half (0/1)
    const int m  = lane & 15;
    const int kb = (lane >> 4) * 8;                // K lane-group offset (0/8)

    const int ldr  = tid >> 2;                     // 0..63: staged row/col
    const int ldc8 = (tid & 3) * 8;                // 0,8,16,24: K chunk of 8 f32

    const float* xg = x + (size_t)(row0 + ldr) * IN_F + ldc8;
    const float* wg = W + (size_t)(col0 + ldr) * IN_F + ldc8;
    float* const xd0 = &xs[0][ldr * LDF + ldc8];
    float* const xd1 = &xs[1][ldr * LDF + ldc8];
    float* const wd0 = &wl[0][ldr * LDF + ldc8];
    float* const wd1 = &wl[1][ldr * LDF + ldc8];

    // Prologue: stage K-tile 0 into buffer 0 (2x b128 for x, 2x for W).
    ALD16(xd0, xg, 0);  ALD16(xd0, xg, 16);
    ALD16(wd0, wg, 0);  ALD16(wd0, wg, 16);

    v8f acc[2] = {v8f{}, v8f{}};
    int buf = 0;
    for (int k0 = 0; k0 < IN_F; k0 += 32, buf ^= 1) {
        if (k0 + 32 < IN_F) {                      // stream next tile into buf^1
            const float* xn = xg + k0 + 32;
            const float* wn = wg + k0 + 32;
            float* xdn = buf ? xd0 : xd1;
            float* wdn = buf ? wd0 : wd1;
            ALD16(xdn, xn, 0);  ALD16(xdn, xn, 16);
            ALD16(wdn, wn, 0);  ALD16(wdn, wn, 16);
            WAIT_ASYNC(4);                         // tile k done (in-order), k+1 in flight
        } else {
            WAIT_ASYNC(0);
        }
        __syncthreads();                           // whole tile visible to all waves

        // A fragment: row mr*16+m of the x tile (f32 -> f16 on read)
        const float* xrow = &xs[buf][(mr * 16 + m) * LDF + kb];
        v16h a;
        #pragma unroll
        for (int j = 0; j < 8; ++j) {
            a[j]     = (_Float16)xrow[j];
            a[j + 8] = (_Float16)xrow[16 + j];
        }
        // B fragments: cols nc*32+m and nc*32+16+m of the W tile
        const float* w0 = &wl[buf][(nc * 32 + m) * LDF + kb];
        const float* w1 = &wl[buf][(nc * 32 + 16 + m) * LDF + kb];
        v16h bf0, bf1;
        #pragma unroll
        for (int j = 0; j < 8; ++j) {
            bf0[j]     = (_Float16)w0[j];
            bf0[j + 8] = (_Float16)w0[16 + j];
            bf1[j]     = (_Float16)w1[j];
            bf1[j + 8] = (_Float16)w1[16 + j];
        }

        acc[0] = __builtin_amdgcn_wmma_f32_16x16x32_f16(false, a, false, bf0,
                                                        (short)0, acc[0], false, false);
        acc[1] = __builtin_amdgcn_wmma_f32_16x16x32_f16(false, a, false, bf1,
                                                        (short)0, acc[1], false, false);
        __syncthreads();                           // reads done before buf refilled
    }

    // Epilogue: rank-16 adapter, exact f32 via v_wmma_f32_16x16x4_f32 + bias.
    const int rowt = row0 + mr * 16;
    const float* __restrict__ Trow = T + (size_t)rowt * RANK;
    const int koff = (lane >= 16) ? 2 : 0;

    #pragma unroll
    for (int ct = 0; ct < 2; ++ct) {
        const int gcol = col0 + nc * 32 + ct * 16 + m;
        const float* __restrict__ Bcol = Bmix + (size_t)b * RANK * OUT_F + gcol;
        #pragma unroll
        for (int c4 = 0; c4 < 4; ++c4) {           // K = 16 in chunks of 4
            const int kk = c4 * 4 + koff;
            v2f af, bf;
            af.x = Trow[m * RANK + kk];
            af.y = Trow[m * RANK + kk + 1];
            bf.x = Bcol[(size_t)kk * OUT_F];
            bf.y = Bcol[(size_t)(kk + 1) * OUT_F];
            acc[ct] = __builtin_amdgcn_wmma_f32_16x16x4_f32(false, af, false, bf,
                                                            (short)0, acc[ct],
                                                            false, false);
        }
        const float bv = bias[gcol];
        #pragma unroll
        for (int j = 0; j < 8; ++j) {
            const int mrow = ((lane < 16) ? 0 : 8) + j;
            out[(size_t)(rowt + mrow) * OUT_F + gcol] = acc[ct][j] + bv;
        }
    }
}

// ---------------------------------------------------------------------------
extern "C" void kernel_launch(void* const* d_in, const int* in_sizes, int n_in,
                              void* d_out, int out_size, void* d_ws, size_t ws_size,
                              hipStream_t stream) {
    const float* x       = (const float*)d_in[0];
    const float* weights = (const float*)d_in[1];
    const float* W       = (const float*)d_in[2];
    const float* bias    = (const float*)d_in[3];
    const float* lora_a  = (const float*)d_in[4];
    const float* lora_b  = (const float*)d_in[5];
    float* out = (float*)d_out;

    float* Amix = (float*)d_ws;            // 262144 f32 = 1 MB
    float* Bmix = Amix + MIXN;             // 262144 f32 = 1 MB
    float* T    = Bmix + MIXN;             // 16384*16 f32 = 1 MB

    skilled_lora_mix<<<MIXN / 256, 256, 0, stream>>>(weights, lora_a, lora_b,
                                                     Amix, Bmix);
    skilled_lora_xa<<<(NROWS / 16) / 4, 128, 0, stream>>>(x, Amix, T);
    skilled_lora_gemm<<<dim3(OUT_F / 64, NROWS / 64), 256, 0, stream>>>(
        x, W, bias, T, Bmix, out);
}
